// AmazonNet_4964982194530
// MI455X (gfx1250) — matrix operand order repaired and checked by reading, not compiled, not run on hardware
//
#include <hip/hip_runtime.h>
#include <hip/hip_bf16.h>
#include <stdint.h>

// ---------------- CDNA5 WMMA types ----------------
typedef __attribute__((ext_vector_type(16))) __bf16       v16bf;
typedef __attribute__((ext_vector_type(8)))  float        v8f;
typedef __attribute__((ext_vector_type(4)))  unsigned int v4u;

union FragAB { v16bf bf; v4u u4[2]; };
static_assert(sizeof(FragAB) == 32, "frag size");

// ---------------- problem constants (match reference) ----------------
constexpr int Hh  = 3;    // heads
constexpr int Cc  = 64;   // channels per head
constexpr int HC  = 192;  // H*C
constexpr int Ff  = 128;  // input features
constexpr int Gg  = 16;   // graphs
constexpr int NCl = 10;   // classes
constexpr float SLOPE = 0.2f;

// ---------------- helpers ----------------
__device__ __forceinline__ unsigned short f2bf(float f) {
  unsigned int u = __float_as_uint(f);
  u = u + 0x7FFFu + ((u >> 16) & 1u);          // round-to-nearest-even
  return (unsigned short)(u >> 16);
}
__device__ __forceinline__ int   f2ord(float f) { int i = __float_as_int(f); return i >= 0 ? i : (i ^ 0x7FFFFFFF); }
__device__ __forceinline__ float ord2f(int i)   { return __int_as_float(i >= 0 ? i : (i ^ 0x7FFFFFFF)); }

// ---------------- init / conversion kernels ----------------
__global__ void k_fill_f32(float* p, float v, int n) {
  int i = blockIdx.x * blockDim.x + threadIdx.x;
  if (i < n) p[i] = v;
}
__global__ void k_fill_i32(int* p, int v, int n) {
  int i = blockIdx.x * blockDim.x + threadIdx.x;
  if (i < n) p[i] = v;
}
__global__ void k_f32_to_bf16(const float* __restrict__ in, unsigned short* __restrict__ out, int n) {
  int i = blockIdx.x * blockDim.x + threadIdx.x;
  if (i < n) out[i] = f2bf(in[i]);
}
// W [K, Nout] fp32 row-major -> Wt [Nout, K] bf16 (transposed so B-fragments load contiguously in K)
__global__ void k_conv_wt(const float* __restrict__ W, unsigned short* __restrict__ Wt, int K, int Nout) {
  int t = blockIdx.x * blockDim.x + threadIdx.x;
  if (t >= K * Nout) return;
  int k = t / Nout, n = t % Nout;
  Wt[n * K + k] = f2bf(W[t]);
}

// ---------------- dual-output WMMA GEMM ------------------------------------------------
// C1 = A @ Bt1^T, C2 = A @ Bt2^T  (A [M,K] bf16, Bt [HC,K] bf16, C [M,HC] fp32)
// One wave per 16-row band of A: the K-strip of A fragments is loaded ONCE into registers
// (KSTEPS*8 VGPRs), then reused across all 12 N-tiles of BOTH weight matrices -> 24x less
// A traffic than tile-per-wave. B (<=73KB) stays resident in L0/L2.
template <int KSTEPS>
__global__ void k_gemm_dual_wmma(const unsigned short* __restrict__ A,
                                 const unsigned short* __restrict__ Bt1,
                                 const unsigned short* __restrict__ Bt2,
                                 float* __restrict__ C1, float* __restrict__ C2,
                                 int M) {
  constexpr int K = KSTEPS * 32;
  const int lane  = threadIdx.x & 31;
  const int wave  = threadIdx.x >> 5;
  const int tileM = blockIdx.x * (blockDim.x >> 5) + wave;
  if (tileM * 16 >= M) return;                 // wave-uniform: EXEC all-ones for WMMA

  const int l15 = lane & 15;
  const int hi  = lane >> 4;                   // lane half
  const int row = tileM * 16 + l15;

  // A 16x32 bf16 fragment layout: lane<16 -> K {kk..kk+7, kk+16..kk+23}; lane>=16 -> +8
  FragAB fa[KSTEPS];
#pragma unroll
  for (int s = 0; s < KSTEPS; ++s) {
    const unsigned short* ap = A + (long long)row * K + s * 32 + hi * 8;
    fa[s].u4[0] = *(const v4u*)(ap);
    fa[s].u4[1] = *(const v4u*)(ap + 16);
  }
  const int m0 = tileM * 16 + hi * 8;

#pragma unroll
  for (int which = 0; which < 2; ++which) {
    const unsigned short* Bt = which ? Bt2 : Bt1;
    float*                C  = which ? C2  : C1;
    for (int tn = 0; tn < HC / 16; ++tn) {
      const int col = tn * 16 + l15;
      v8f acc = {};
#pragma unroll
      for (int s = 0; s < KSTEPS; ++s) {
        FragAB fb;                              // B 32x16: lane<16 K {..+7, +16..+23}, lane>=16 -> +8
        const unsigned short* bp = Bt + (long long)col * K + s * 32 + hi * 8;
        fb.u4[0] = *(const v4u*)(bp);
        fb.u4[1] = *(const v4u*)(bp + 16);
        acc = __builtin_amdgcn_wmma_f32_16x16x32_bf16(false, fa[s].bf, false, fb.bf,
                                                      (short)0, acc, false, false);
      }
      const int n0 = tn * 16 + l15;
#pragma unroll
      for (int r = 0; r < 8; ++r)
        C[(long long)(m0 + r) * HC + n0] = acc[r];
    }
  }
}

// ---------------- edge-attention passes (segment softmax over dst) ----------------
// Pass A, wave-per-edge: lane l covers channels {2l,2l+1} of each head -> each head row is
// one contiguous 256B wave request (coalesced). 5-step shfl_xor reduction per head.
__global__ void k_edge_logits_max(const float* __restrict__ xl, const float* __restrict__ xr,
                                  const float* __restrict__ att,
                                  const int* __restrict__ esrc, const int* __restrict__ edst,
                                  int E, int N,
                                  float* __restrict__ logits, int* __restrict__ mord) {
  int e    = (blockIdx.x * blockDim.x + threadIdx.x) >> 5;
  int lane = threadIdx.x & 31;
  if (e >= E + N) return;
  int s, d;
  if (e < E) { s = esrc[e]; d = edst[e]; } else { s = d = e - E; }   // appended self-loops
  const float2* pl = (const float2*)(xl + (long long)s * HC);
  const float2* pr = (const float2*)(xr + (long long)d * HC);
  const float2* pa = (const float2*)att;
  float r[Hh];
#pragma unroll
  for (int h = 0; h < Hh; ++h) {
    int idx = h * 32 + lane;                    // float2 index inside the 192-wide row
    float2 a = pl[idx], b = pr[idx], w = pa[idx];
    float v0 = a.x + b.x; v0 = v0 > 0.f ? v0 : SLOPE * v0;
    float v1 = a.y + b.y; v1 = v1 > 0.f ? v1 : SLOPE * v1;
    float p = v0 * w.x + v1 * w.y;
#pragma unroll
    for (int off = 16; off; off >>= 1) p += __shfl_xor(p, off, 32);
    r[h] = p;
  }
  if (lane == 0) {
#pragma unroll
    for (int h = 0; h < Hh; ++h) {
      logits[e * Hh + h] = r[h];
      atomicMax(&mord[d * Hh + h], f2ord(r[h]));
    }
  }
}

// Pass B: p = exp(logit - max[dst]); denom[dst] += p  (p overwrites logits). Light, coalesced.
__global__ void k_edge_exp_denom(const int* __restrict__ edst, int E, int N,
                                 float* __restrict__ logits, const int* __restrict__ mord,
                                 float* __restrict__ denom) {
  int t = blockIdx.x * blockDim.x + threadIdx.x;
  int total = (E + N) * Hh;
  if (t >= total) return;
  int h = t % Hh, e = t / Hh;
  int d = (e < E) ? edst[e] : (e - E);
  float p = __expf(logits[e * Hh + h] - ord2f(mord[d * Hh + h]));
  logits[e * Hh + h] = p;
  atomicAdd(&denom[d * Hh + h], p);
}

// Pass C, wave-per-edge: accum[dst] += alpha_h * xl[src]; 2 coalesced atomics per lane per head.
__global__ void k_edge_scatter(const float* __restrict__ xl,
                               const float* __restrict__ logits, const float* __restrict__ denom,
                               const int* __restrict__ esrc, const int* __restrict__ edst,
                               int E, int N, float* __restrict__ accum) {
  int e    = (blockIdx.x * blockDim.x + threadIdx.x) >> 5;
  int lane = threadIdx.x & 31;
  if (e >= E + N) return;
  int s, d;
  if (e < E) { s = esrc[e]; d = edst[e]; } else { s = d = e - E; }
  const float2* pl = (const float2*)(xl + (long long)s * HC);
#pragma unroll
  for (int h = 0; h < Hh; ++h) {
    float alpha = logits[e * Hh + h] / fmaxf(denom[d * Hh + h], 1e-16f);  // broadcast loads
    float2 v = pl[h * 32 + lane];
    float* po = accum + (long long)d * HC + h * Cc + 2 * lane;
    atomicAdd(po + 0, alpha * v.x);
    atomicAdd(po + 1, alpha * v.y);
  }
}

// layer-1 epilogue: h = relu(accum + b1), re-quantized to bf16 for the next WMMA GEMM
__global__ void k_bias_relu_bf16(const float* __restrict__ accum, const float* __restrict__ b,
                                 unsigned short* __restrict__ out, int n) {
  int i = blockIdx.x * blockDim.x + threadIdx.x;
  if (i >= n) return;
  float v = accum[i] + b[i % HC];
  out[i] = f2bf(v > 0.f ? v : 0.f);
}

// layer-2 epilogue + global mean pool: v = relu(mean_heads(accum)+b2); gsum[batch] += v
__global__ void k_mean_relu_pool(const float* __restrict__ accum, const float* __restrict__ b2,
                                 const int* __restrict__ batch,
                                 float* __restrict__ gsum, float* __restrict__ gcnt, int N) {
  int t = blockIdx.x * blockDim.x + threadIdx.x;
  if (t >= N * Cc) return;
  int n = t / Cc, c = t % Cc;
  const float* a = accum + (long long)n * HC;
  float v = (a[c] + a[Cc + c] + a[2 * Cc + c]) * (1.0f / 3.0f) + b2[c];
  v = v > 0.f ? v : 0.f;
  int g = batch[n];
  atomicAdd(&gsum[g * Cc + c], v);
  if (c == 0) atomicAdd(&gcnt[g], 1.0f);
}

// classifier: softmax((gsum/cnt) @ Wc + bc); one thread per graph (16 total — trivial)
__global__ void k_classify(const float* __restrict__ gsum, const float* __restrict__ gcnt,
                           const float* __restrict__ Wc, const float* __restrict__ bc,
                           float* __restrict__ out) {
  int g = threadIdx.x;
  if (g >= Gg) return;
  float cnt = fmaxf(gcnt[g], 1.0f);
  float lg[NCl], mx = -INFINITY;
#pragma unroll
  for (int k = 0; k < NCl; ++k) {
    float a = bc[k];
    for (int c = 0; c < Cc; ++c) a += (gsum[g * Cc + c] / cnt) * Wc[c * NCl + k];
    lg[k] = a;
    mx = a > mx ? a : mx;
  }
  float s = 0.f;
#pragma unroll
  for (int k = 0; k < NCl; ++k) { lg[k] = __expf(lg[k] - mx); s += lg[k]; }
#pragma unroll
  for (int k = 0; k < NCl; ++k) out[g * NCl + k] = lg[k] / s;
}

// ---------------- launcher ----------------
extern "C" void kernel_launch(void* const* d_in, const int* in_sizes, int n_in,
                              void* d_out, int out_size, void* d_ws, size_t ws_size,
                              hipStream_t stream) {
  const float* x    = (const float*)d_in[0];
  const int*   eidx = (const int*)  d_in[1];
  const int*   batch= (const int*)  d_in[2];
  const float* Wl1  = (const float*)d_in[3];
  const float* Wr1  = (const float*)d_in[4];
  const float* att1 = (const float*)d_in[5];
  const float* b1   = (const float*)d_in[6];
  const float* Wl2  = (const float*)d_in[7];
  const float* Wr2  = (const float*)d_in[8];
  const float* att2 = (const float*)d_in[9];
  const float* b2   = (const float*)d_in[10];
  const float* Wc   = (const float*)d_in[11];
  const float* bc   = (const float*)d_in[12];
  float* out = (float*)d_out;

  const int N = in_sizes[0] / Ff;
  const int E = in_sizes[1] / 2;
  const int* esrc = eidx;
  const int* edst = eidx + E;

  // workspace carving (256B aligned)
  char* ws = (char*)d_ws;
  size_t off = 0;
  auto carve = [&](size_t bytes) -> char* {
    char* p = ws + off;
    off += (bytes + 255) & ~(size_t)255;
    return p;
  };
  unsigned short* xbf   = (unsigned short*)carve((size_t)N * Ff * 2);
  unsigned short* wl1t  = (unsigned short*)carve((size_t)HC * Ff * 2);
  unsigned short* wr1t  = (unsigned short*)carve((size_t)HC * Ff * 2);
  unsigned short* wl2t  = (unsigned short*)carve((size_t)HC * HC * 2);
  unsigned short* wr2t  = (unsigned short*)carve((size_t)HC * HC * 2);
  float*          xl    = (float*)carve((size_t)N * HC * 4);
  float*          xr    = (float*)carve((size_t)N * HC * 4);
  float*          logit = (float*)carve((size_t)(E + N) * Hh * 4);
  int*            mord  = (int*)  carve((size_t)N * Hh * 4);
  float*          denom = (float*)carve((size_t)N * Hh * 4);
  float*          accum = (float*)carve((size_t)N * HC * 4);
  unsigned short* hbf   = (unsigned short*)carve((size_t)N * HC * 2);
  float*          gsum  = (float*)carve((size_t)Gg * Cc * 4);
  float*          gcnt  = (float*)carve((size_t)Gg * 4);
  (void)ws_size; (void)n_in; (void)out_size;

  const int TB = 256;
  auto nb = [](long long n, int tb) { return (unsigned)((n + tb - 1) / tb); };

  // ---- conversions ----
  k_f32_to_bf16<<<nb((long long)N * Ff, TB), TB, 0, stream>>>(x, xbf, N * Ff);
  k_conv_wt<<<nb(Ff * HC, TB), TB, 0, stream>>>(Wl1, wl1t, Ff, HC);
  k_conv_wt<<<nb(Ff * HC, TB), TB, 0, stream>>>(Wr1, wr1t, Ff, HC);
  k_conv_wt<<<nb(HC * HC, TB), TB, 0, stream>>>(Wl2, wl2t, HC, HC);
  k_conv_wt<<<nb(HC * HC, TB), TB, 0, stream>>>(Wr2, wr2t, HC, HC);

  const int tilesM = (N + 15) / 16;            // 3125 (N divisible by 16)
  const unsigned gblocks = nb((long long)tilesM * 32, TB);   // 8 waves/block, 1 row-band/wave

  const long long EA = (long long)E + N;
  const int INT_MINV = (int)0x80000000;

  // ---- layer 1 ----
  k_gemm_dual_wmma<Ff / 32><<<gblocks, TB, 0, stream>>>(xbf, wl1t, wr1t, xl, xr, N);
  k_fill_i32<<<nb(N * Hh, TB), TB, 0, stream>>>(mord, INT_MINV, N * Hh);
  k_fill_f32<<<nb(N * Hh, TB), TB, 0, stream>>>(denom, 0.f, N * Hh);
  k_fill_f32<<<nb((long long)N * HC, TB), TB, 0, stream>>>(accum, 0.f, N * HC);
  k_edge_logits_max<<<nb(EA * 32, TB), TB, 0, stream>>>(xl, xr, att1, esrc, edst, E, N, logit, mord);
  k_edge_exp_denom <<<nb(EA * Hh, TB), TB, 0, stream>>>(edst, E, N, logit, mord, denom);
  k_edge_scatter   <<<nb(EA * 32, TB), TB, 0, stream>>>(xl, logit, denom, esrc, edst, E, N, accum);
  k_bias_relu_bf16 <<<nb((long long)N * HC, TB), TB, 0, stream>>>(accum, b1, hbf, N * HC);

  // ---- layer 2 ----
  k_gemm_dual_wmma<HC / 32><<<gblocks, TB, 0, stream>>>(hbf, wl2t, wr2t, xl, xr, N);
  k_fill_i32<<<nb(N * Hh, TB), TB, 0, stream>>>(mord, INT_MINV, N * Hh);
  k_fill_f32<<<nb(N * Hh, TB), TB, 0, stream>>>(denom, 0.f, N * Hh);
  k_fill_f32<<<nb((long long)N * HC, TB), TB, 0, stream>>>(accum, 0.f, N * HC);
  k_edge_logits_max<<<nb(EA * 32, TB), TB, 0, stream>>>(xl, xr, att2, esrc, edst, E, N, logit, mord);
  k_edge_exp_denom <<<nb(EA * Hh, TB), TB, 0, stream>>>(edst, E, N, logit, mord, denom);
  k_edge_scatter   <<<nb(EA * 32, TB), TB, 0, stream>>>(xl, logit, denom, esrc, edst, E, N, accum);

  // ---- pool + classify ----
  k_fill_f32<<<nb(Gg * Cc, TB), TB, 0, stream>>>(gsum, 0.f, Gg * Cc);
  k_fill_f32<<<1, 32, 0, stream>>>(gcnt, 0.f, Gg);
  k_mean_relu_pool<<<nb((long long)N * Cc, TB), TB, 0, stream>>>(accum, b2, batch, gsum, gcnt, N);
  k_classify<<<1, 32, 0, stream>>>(gsum, gcnt, Wc, bc, out);
}
